// GATE_GRUConv_IntraMol_55516747268874
// MI455X (gfx1250) — compile-verified, matrix operand order checked by reading, not run on hardware
//
#include <hip/hip_runtime.h>
#include <hip/hip_bf16.h>
#include <math.h>

// ===========================================================================
// GVP-GNN layer for MI455X (gfx1250).  All heavy GEMMs run on
// v_wmma_f32_16x16x32_f16 (f16 in, f32 accumulate).  The per-edge message
// pipeline (gaussian smearing -> edge_gvl -> sca_W -> msg.out -> elu ->
// segment-sum) is fused into one kernel so no per-edge intermediate touches
// HBM.  Both WMMA operands are kept in fragment-native swizzled LDS layout:
// every operand load is 2x ds_load_b128.  Full tiles take a wave-uniform
// fast path (no per-row EXEC masking).
//
// Input flattening assumption (setup_inputs() insertion order, recursive):
//  0 x_sca  1 pos  2 edge_index  3 edge_attr, then params leaves in dict
//  insertion order (see enum below).
// ===========================================================================

typedef __attribute__((ext_vector_type(16))) _Float16 v16h;
typedef __attribute__((ext_vector_type(4)))  _Float16 v4h;
typedef __attribute__((ext_vector_type(8)))  float    v8f;

#define LEAKY_S 0.01f
#define LEAKY_V 0.2f
#define VN_EPS  1e-6f
#define PI_F    3.14159265358979f

// ---------------------------------------------------------------- WMMA utils
// 16-bit A-matrix 16x32 layout (ISA 7.12.2): lane l holds row M=l&15;
// lanes 0-15 cover K={0..7,16..23}, lanes 16-31 cover K={8..15,24..31}.
// Swizzled element offset for matrix element (m, k) inside a fragment-native
// buffer laid out as [kblock][lane][16 halves]:
__device__ __forceinline__ int swzA(int m, int k) {
  int kb = k >> 5, k32 = k & 31;
  int hi = (k32 >> 3) & 1;
  int h = (k32 & 7) + ((k32 & 16) ? 8 : 0);
  return (kb << 9) + ((m + (hi << 4)) << 4) + h;
}
// A fragment: contiguous 32B per lane (2x ds_load_b128)
__device__ __forceinline__ v16h fragAs(const _Float16* t, int kb, int lane) {
  return ((const v16h*)t)[(kb << 5) + lane];
}
// B fragment from fragment-packed weights [kb][nb][lane][16]
__device__ __forceinline__ v16h fragBs(const _Float16* w, int NB, int kb,
                                       int nb, int lane) {
  return ((const v16h*)w)[((kb * NB + nb) << 5) + lane];
}
__device__ __forceinline__ v8f wmma16(v16h a, v16h b, v8f c) {
  return __builtin_amdgcn_wmma_f32_16x16x32_f16(false, a, false, b, (short)0, c,
                                                false, false);
}
__device__ __forceinline__ float sigm(float x) { return 1.f / (1.f + __expf(-x)); }
__device__ __forceinline__ float eluf(float x) { return x > 0.f ? x : (__expf(x) - 1.f); }
__device__ __forceinline__ float lrelu(float x, float s) { return x > 0.f ? x : s * x; }
// VNLeakyReLU with 1x1 weight u acting on a single 3-vector (in place)
__device__ __forceinline__ void vnleaky1(float u, float& x0, float& x1, float& x2) {
  float xx = x0 * x0 + x1 * x1 + x2 * x2;
  float dot = u * xx;
  float denom = u * u * xx + VN_EPS;
  float projf = 1.f - (u * u * xx) / denom;
  float f = LEAKY_V + (1.f - LEAKY_V) * ((dot >= 0.f) ? 1.f : projf);
  x0 *= f; x1 *= f; x2 *= f;
}

// ------------------------------------------------------------ weight packing
// src W row-major (N, Kd+has_vn) f32.  dst f16, fragment-native:
//   frag region: KB*NB*512 halves, [kb][nb][lane][h],
//     element = W[nb*16 + (lane&15)][kmap(lane,h)+has_vn]
//   then (if has_vn) N halves: vnorm column W[n][0]
__global__ void pack_wt_kernel(const float* W, _Float16* dst, int N, int Kd,
                               int has_vn) {
  int KB = Kd >> 5, NB = N >> 4;
  int fragTotal = KB * NB * 512;
  int total = fragTotal + (has_vn ? N : 0);
  int Ksrc = Kd + has_vn;
  for (int i = blockIdx.x * blockDim.x + threadIdx.x; i < total;
       i += gridDim.x * blockDim.x) {
    if (i < fragTotal) {
      int h = i & 15, lane = (i >> 4) & 31;
      int blk = i >> 9;
      int nb = blk % NB, kb = blk / NB;
      int k32 = ((lane & 16) ? 8 : 0) + (h < 8 ? h : h + 8);
      int k = kb * 32 + k32;
      int n = nb * 16 + (lane & 15);
      dst[i] = (_Float16)W[(size_t)n * Ksrc + k + has_vn];
    } else {
      int n = i - fragTotal;
      dst[i] = (_Float16)W[(size_t)n * Ksrc];
    }
  }
}

__global__ void zero_kernel(float* p, long long cnt) {
  for (long long i = blockIdx.x * (long long)blockDim.x + threadIdx.x; i < cnt;
       i += (long long)gridDim.x * blockDim.x)
    p[i] = 0.f;
}

// --------------------------------------------------- node gv_lin GEMM (WMMA)
// out_pre[node][n] = bias[n] + vnorm(node)*W[n][0] + sum_k in_sca[node][k]*W[n][k+1]
#define NW 4
__global__ void gvlin_gemm_kernel(const float* in_sca, const float* in_vec,
                                  const float* w1p, const _Float16* Wt,
                                  const float* bias, float* out_pre, int n,
                                  int nTiles) {
  __shared__ __align__(32) _Float16 sW[4 * 8 * 512 + 128];  // frags + vn row
  __shared__ __align__(32) _Float16 sA[NW][2048];           // swizzled A tile
  __shared__ float sVn[NW][16];
  __shared__ float sB[128];
  int tid = threadIdx.x, lane = tid & 31, w = tid >> 5;
  for (int i = tid; i < 4 * 8 * 512 + 128; i += blockDim.x) sW[i] = Wt[i];
  for (int i = tid; i < 128; i += blockDim.x) sB[i] = bias[i];
  __syncthreads();
  float w1a = fabsf(w1p[0]);
  _Float16* A = sA[w];
  int ncb = lane & 15, rowhi = (lane >> 4) * 8;
  for (int tb = blockIdx.x; tb * NW < nTiles; tb += gridDim.x) {
    int t = tb * NW + w;
    bool act = t < nTiles;
    int base = t * 16;
    bool full = act && (base + 16 <= n);
    if (act) {
      if (full) {  // fast path: no per-row bounds checks
        for (int i = lane; i < 16 * 32; i += 32) {
          int r = i >> 5, q = i & 31;
          float4 v = ((const float4*)(in_sca + (size_t)(base + r) * 128))[q];
          v4h hv = {(_Float16)v.x, (_Float16)v.y, (_Float16)v.z, (_Float16)v.w};
          *(v4h*)(A + swzA(r, q * 4)) = hv;
        }
        if (lane < 16) {
          int node = base + lane;
          float vx = in_vec[(size_t)node * 3 + 0];
          float vy = in_vec[(size_t)node * 3 + 1];
          float vz = in_vec[(size_t)node * 3 + 2];
          sVn[w][lane] = w1a * sqrtf(vx * vx + vy * vy + vz * vz);
        }
      } else {
        for (int i = lane; i < 16 * 32; i += 32) {
          int r = i >> 5, q = i & 31;
          int node = base + r;
          float4 v = make_float4(0.f, 0.f, 0.f, 0.f);
          if (node < n) v = ((const float4*)(in_sca + (size_t)node * 128))[q];
          v4h hv = {(_Float16)v.x, (_Float16)v.y, (_Float16)v.z, (_Float16)v.w};
          *(v4h*)(A + swzA(r, q * 4)) = hv;
        }
        if (lane < 16) {
          int node = base + lane;
          float vx = 0.f, vy = 0.f, vz = 0.f;
          if (node < n) {
            vx = in_vec[(size_t)node * 3 + 0];
            vy = in_vec[(size_t)node * 3 + 1];
            vz = in_vec[(size_t)node * 3 + 2];
          }
          sVn[w][lane] = w1a * sqrtf(vx * vx + vy * vy + vz * vz);
        }
      }
    }
    __syncthreads();
    if (act) {
      float vnR[8];
#pragma unroll
      for (int r = 0; r < 8; ++r) vnR[r] = sVn[w][rowhi + r];
      for (int nt = 0; nt < 8; ++nt) {
        v8f acc = {0.f, 0.f, 0.f, 0.f, 0.f, 0.f, 0.f, 0.f};
#pragma unroll
        for (int kb = 0; kb < 4; ++kb)
          acc = wmma16(fragAs(A, kb, lane), fragBs(sW, 8, kb, nt, lane), acc);
        int ncol = nt * 16 + ncb;
        float wv = (float)sW[4 * 8 * 512 + ncol];
        float bb = sB[ncol];
        if (full) {
#pragma unroll
          for (int r = 0; r < 8; ++r)
            out_pre[(size_t)(base + rowhi + r) * 128 + ncol] =
                acc[r] + bb + vnR[r] * wv;
        } else {
#pragma unroll
          for (int r = 0; r < 8; ++r) {
            int node = base + rowhi + r;
            if (node < n)
              out_pre[(size_t)node * 128 + ncol] = acc[r] + bb + vnR[r] * wv;
          }
        }
      }
    }
    __syncthreads();
  }
}

// --------------------------------------------- node gv_lin epilogue (gates)
// mode 0: plain gv_lin.  mode 1: gv_perceptron (leaky scalar + vn_leaky vec)
__global__ void gvlin_epi_kernel(const float* pre, const float* in_vec,
                                 const float* w1p, const float* w2p,
                                 const float* gW, const float* gb,
                                 const float* vnWp, int mode, float* out_sca,
                                 float* out_vec, int n) {
  int node = blockIdx.x;
  if (node >= n) return;
  int f = threadIdx.x, lane = f & 31, wv = f >> 5;
  float s = pre[(size_t)node * 128 + f];
  float p = s * gW[f];
  p += __shfl_xor(p, 1);  p += __shfl_xor(p, 2);
  p += __shfl_xor(p, 4);  p += __shfl_xor(p, 8);  p += __shfl_xor(p, 16);
  __shared__ float red[4];
  if (lane == 0) red[wv] = p;
  __syncthreads();
  float gate = sigm(red[0] + red[1] + red[2] + red[3] + gb[0]);
  out_sca[(size_t)node * 128 + f] = (mode == 1) ? lrelu(s, LEAKY_S) : s;
  if (f == 0) {
    float w1 = w1p[0], w2 = w2p[0];
    float x0 = gate * w2 * w1 * in_vec[(size_t)node * 3 + 0];
    float x1 = gate * w2 * w1 * in_vec[(size_t)node * 3 + 1];
    float x2 = gate * w2 * w1 * in_vec[(size_t)node * 3 + 2];
    if (mode == 1) vnleaky1(vnWp[0], x0, x1, x2);
    out_vec[(size_t)node * 3 + 0] = x0;
    out_vec[(size_t)node * 3 + 1] = x1;
    out_vec[(size_t)node * 3 + 2] = x2;
  }
}

// --------------------------------------------------- LayerNorm + activations
__global__ void norm_act_kernel(const float* cs, const float* cv,
                                const float* asca, const float* avec,
                                const float* lnsw, const float* lnsb,
                                const float* lnvw, const float* lnvb,
                                const float* actWp, float* lns, float* lnv,
                                int n) {
  int node = blockIdx.x;
  if (node >= n) return;
  int f = threadIdx.x, lane = f & 31, wv = f >> 5;
  float x = cs[(size_t)node * 128 + f] + asca[(size_t)node * 128 + f];
  float s = x, s2 = x * x;
  s  += __shfl_xor(s, 1);  s2 += __shfl_xor(s2, 1);
  s  += __shfl_xor(s, 2);  s2 += __shfl_xor(s2, 2);
  s  += __shfl_xor(s, 4);  s2 += __shfl_xor(s2, 4);
  s  += __shfl_xor(s, 8);  s2 += __shfl_xor(s2, 8);
  s  += __shfl_xor(s, 16); s2 += __shfl_xor(s2, 16);
  __shared__ float rs[4], rq[4];
  if (lane == 0) { rs[wv] = s; rq[wv] = s2; }
  __syncthreads();
  float sum = rs[0] + rs[1] + rs[2] + rs[3];
  float sq  = rq[0] + rq[1] + rq[2] + rq[3];
  float mu = sum * (1.f / 128.f);
  float var = sq * (1.f / 128.f) - mu * mu;
  float y = (x - mu) * rsqrtf(var + 1e-5f) * lnsw[f] + lnsb[f];
  lns[(size_t)node * 128 + f] = lrelu(y, LEAKY_S);
  if (f == 0) {
    float v0 = cv[(size_t)node * 3 + 0] + avec[(size_t)node * 3 + 0];
    float v1 = cv[(size_t)node * 3 + 1] + avec[(size_t)node * 3 + 1];
    float v2 = cv[(size_t)node * 3 + 2] + avec[(size_t)node * 3 + 2];
    float m3 = (v0 + v1 + v2) * (1.f / 3.f);
    float va = ((v0 - m3) * (v0 - m3) + (v1 - m3) * (v1 - m3) +
                (v2 - m3) * (v2 - m3)) * (1.f / 3.f);
    float inv = rsqrtf(va + 1e-5f);
    float t0 = (v0 - m3) * inv * lnvw[0] + lnvb[0];
    float t1 = (v1 - m3) * inv * lnvw[1] + lnvb[1];
    float t2 = (v2 - m3) * inv * lnvw[2] + lnvb[2];
    vnleaky1(actWp[0], t0, t1, t2);
    lnv[(size_t)node * 3 + 0] = t0;
    lnv[(size_t)node * 3 + 1] = t1;
    lnv[(size_t)node * 3 + 2] = t2;
  }
}

// ------------------------------------------------ fused per-edge message kernel
// One wave processes a tile of 16 edges; 3 chained WMMA GEMMs stay in LDS/regs.
#define EW 2
__global__ void __launch_bounds__(64) edge_fused_kernel(
    const float* pos, const float* eattr, const int* eidx,
    const float* ns_node, const float* nv_node, const _Float16* WtE1,
    const _Float16* WtSca, const _Float16* WtOut, const float* gatesE,
    const float* gEb, const float* b1v, const float* scab, const float* e2nW,
    const float* e2nb_p, const float* n2eW, const float* n2eb_p,
    const float* gatesM, const float* gMb, const float* outb,
    const float* w1e_p, const float* w2e_p, const float* ue_p,
    const float* evnw_p, const float* w1o_p, const float* w2o_p,
    const float* vexp_p, float* a_sca, float* a_vec, int nEdges, int nTiles) {
  __shared__ __align__(32) _Float16 sW1[2 * 4 * 512 + 64];   // edge_gvl + vn row
  __shared__ __align__(32) _Float16 sWs[2 * 8 * 512];        // sca_W
  __shared__ __align__(32) _Float16 sWo[4 * 8 * 512 + 128];  // msg.out + vn row
  __shared__ float cst[64 * 3 + 128 * 4];
  __shared__ __align__(32) _Float16 bufA[EW][2048];  // swz: edge_sca then y_sca
  __shared__ __align__(32) _Float16 bufB[EW][1024];  // swz: es
  __shared__ __align__(32) _Float16 nsb[EW][16 * 132];  // gathered ns, row-major
  __shared__ float sVn[EW][16], sCut[EW][16], sGate[EW][16];
  __shared__ float sAl[EW][16], sBe[EW][16], sVny[EW][16];
  __shared__ float sEvv[EW][16][3], sYv[EW][16][3];
  __shared__ int sRow[EW][16];
  __shared__ int sOk[EW][16];

  int tid = threadIdx.x, lane = tid & 31, w = tid >> 5;
  for (int i = tid; i < 2 * 4 * 512 + 64; i += blockDim.x) sW1[i] = WtE1[i];
  for (int i = tid; i < 2 * 8 * 512; i += blockDim.x) sWs[i] = WtSca[i];
  for (int i = tid; i < 4 * 8 * 512 + 128; i += blockDim.x) sWo[i] = WtOut[i];
  for (int i = tid; i < 64; i += blockDim.x) {
    cst[i] = gatesE[i]; cst[64 + i] = b1v[i]; cst[128 + i] = e2nW[i];
  }
  for (int i = tid; i < 128; i += blockDim.x) {
    cst[192 + i] = scab[i]; cst[320 + i] = n2eW[i];
    cst[448 + i] = gatesM[i]; cst[576 + i] = outb[i];
  }
  __syncthreads();
  const float* cGE = cst;        const float* cB1 = cst + 64;
  const float* cE2N = cst + 128; const float* cSB = cst + 192;
  const float* cN2E = cst + 320; const float* cGM = cst + 448;
  const float* cOB = cst + 576;

  float w1e = w1e_p[0], w2e = w2e_p[0], ue = ue_p[0], evnw = evnw_p[0];
  float w1o = w1o_p[0], w2o = w2o_p[0], vexp = vexp_p[0];
  float geb = gEb[0], gmb = gMb[0], e2nb = e2nb_p[0], n2eb = n2eb_p[0];

  _Float16* A = bufA[w];
  _Float16* B2 = bufB[w];
  _Float16* NS = nsb[w];
  int e0 = lane & 15, hh = lane >> 4, ncb = lane & 15, rowhi = hh * 8;

  const float gdelta = 10.f / 31.f;
  const float gcoef = -0.5f / (gdelta * gdelta);

  for (int tb = blockIdx.x; tb * EW < nTiles; tb += gridDim.x) {
    int t = tb * EW + w;
    bool act = t < nTiles;
    if (act) {
      int base = t * 16;
      bool full = (base + 16 <= nEdges);
      int e = base + e0;
      bool okL = e < nEdges;
      int ec = okL ? e : (nEdges - 1);
      int rIdx = eidx[ec], cIdx = eidx[nEdges + ec];
      float ex = pos[(size_t)rIdx * 3 + 0] - pos[(size_t)cIdx * 3 + 0];
      float ey = pos[(size_t)rIdx * 3 + 1] - pos[(size_t)cIdx * 3 + 1];
      float ez = pos[(size_t)rIdx * 3 + 2] - pos[(size_t)cIdx * 3 + 2];
      float ed = sqrtf(ex * ex + ey * ey + ez * ez);
      float inv = 1.f / (ed + 1e-7f);
      float vx = vexp * ex * inv, vy = vexp * ey * inv, vz = vexp * ez * inv;
      float vn = fabsf(w1e) * sqrtf(vx * vx + vy * vy + vz * vz);
      float cf = (okL && ed <= 10.f) ? 0.5f * (cosf(ed * (PI_F / 10.f)) + 1.f) : 0.f;
      if (lane < 16) {
        sRow[w][e0] = rIdx; sOk[w][e0] = okL ? 1 : 0;
        sVn[w][e0] = vn;    sCut[w][e0] = cf;
        sEvv[w][e0][0] = vx; sEvv[w][e0][1] = vy; sEvv[w][e0][2] = vz; // edge_vec
        sYv[w][e0][0] = nv_node[(size_t)cIdx * 3 + 0];                // nv stash
        sYv[w][e0][1] = nv_node[(size_t)cIdx * 3 + 1];
        sYv[w][e0][2] = nv_node[(size_t)cIdx * 3 + 2];
      }
      // stage gaussians + edge_attr into swizzled A (k = 0..63), 4-wide stores
#pragma unroll
      for (int j4 = 0; j4 < 4; ++j4) {
        int colj = hh * 16 + j4 * 4;  // gaussian cols 0..31
        v4h g;
#pragma unroll
        for (int u = 0; u < 4; ++u) {
          float d = ed - gdelta * (colj + u);
          g[u] = (_Float16)__expf(gcoef * d * d);
        }
        *(v4h*)(A + swzA(e0, colj)) = g;
        const float4 av = *(const float4*)(eattr + (size_t)ec * 32 + colj);
        v4h ah = {(_Float16)av.x, (_Float16)av.y, (_Float16)av.z, (_Float16)av.w};
        *(v4h*)(A + swzA(e0, 32 + colj)) = ah;  // attr cols 32..63
      }
      // gather ns[col] -> NS (f16 row-major) and accumulate beta = ns . n2e
      {
        const float4* src = (const float4*)(ns_node + (size_t)cIdx * 128);
        float bp = 0.f;
#pragma unroll
        for (int q = 0; q < 16; ++q) {
          int q4 = hh * 16 + q;
          float4 v = src[q4];
          v4h hv = {(_Float16)v.x, (_Float16)v.y, (_Float16)v.z, (_Float16)v.w};
          *(v4h*)(NS + e0 * 132 + q4 * 4) = hv;
          int ni = q4 * 4;
          bp += v.x * cN2E[ni] + v.y * cN2E[ni + 1] + v.z * cN2E[ni + 2] +
                v.w * cN2E[ni + 3];
        }
        bp += __shfl_xor(bp, 16);
        if (lane < 16) sBe[w][e0] = bp + n2eb;
      }
      // ---- GEMM1: edge_gvl lin_scalar, (16x65)@(65x64), + gate + leaky -> es
      float vnR[8];
#pragma unroll
      for (int r8 = 0; r8 < 8; ++r8) vnR[r8] = sVn[w][rowhi + r8];
      float gp[8] = {0.f, 0.f, 0.f, 0.f, 0.f, 0.f, 0.f, 0.f};
      for (int nt = 0; nt < 4; ++nt) {
        v8f acc = {0.f, 0.f, 0.f, 0.f, 0.f, 0.f, 0.f, 0.f};
#pragma unroll
        for (int kb = 0; kb < 2; ++kb)
          acc = wmma16(fragAs(A, kb, lane), fragBs(sW1, 4, kb, nt, lane), acc);
        int ncol = nt * 16 + ncb;
        float wv = (float)sW1[2 * 4 * 512 + ncol];
        float bb = cB1[ncol], gw = cGE[ncol];
#pragma unroll
        for (int r8 = 0; r8 < 8; ++r8) {
          float val = acc[r8] + bb + vnR[r8] * wv;
          gp[r8] += val * gw;
          B2[swzA(rowhi + r8, ncol)] = (_Float16)lrelu(val, LEAKY_S);
        }
      }
#pragma unroll
      for (int r8 = 0; r8 < 8; ++r8) {
        float v = gp[r8];
        v += __shfl_xor(v, 1); v += __shfl_xor(v, 2);
        v += __shfl_xor(v, 4); v += __shfl_xor(v, 8);
        if (ncb == 0) sGate[w][rowhi + r8] = sigm(v + geb);
      }
      // evv = vn_leaky(gated edge_vec)
      {
        float g = sGate[w][e0];
        float x0 = g * w2e * sEvv[w][e0][0];
        float x1 = g * w2e * sEvv[w][e0][1];
        float x2 = g * w2e * sEvv[w][e0][2];
        vnleaky1(ue, x0, x1, x2);
        if (lane < 16) { sEvv[w][e0][0] = x0; sEvv[w][e0][1] = x1; sEvv[w][e0][2] = x2; }
      }
      // alpha = es . e2n
      {
        float s = 0.f;
        for (int j = 0; j < 32; ++j) {
          int ncol = hh * 32 + j;
          s += (float)B2[swzA(e0, ncol)] * cE2N[ncol];
        }
        s += __shfl_xor(s, 16);
        if (lane < 16) sAl[w][e0] = s + e2nb;
      }
      // ---- GEMM2: y_mult = es @ sca_W^T + b;  y_sca = ns * y_mult -> A (swz)
      for (int nt = 0; nt < 8; ++nt) {
        v8f acc = {0.f, 0.f, 0.f, 0.f, 0.f, 0.f, 0.f, 0.f};
#pragma unroll
        for (int kb = 0; kb < 2; ++kb)
          acc = wmma16(fragAs(B2, kb, lane), fragBs(sWs, 8, kb, nt, lane), acc);
        int ncol = nt * 16 + ncb;
        float bb = cSB[ncol];
#pragma unroll
        for (int r8 = 0; r8 < 8; ++r8) {
          int m = rowhi + r8;
          float y = (float)NS[m * 132 + ncol] * (acc[r8] + bb);
          A[swzA(m, ncol)] = (_Float16)y;
        }
      }
      // y_vec = alpha*nv + beta*edge_vn_W*evv ; vnorm for msg.out
      {
        float al = sAl[w][e0], be = sBe[w][e0];
        float y0 = al * sYv[w][e0][0] + be * evnw * sEvv[w][e0][0];
        float y1 = al * sYv[w][e0][1] + be * evnw * sEvv[w][e0][1];
        float y2 = al * sYv[w][e0][2] + be * evnw * sEvv[w][e0][2];
        float vny = fabsf(w1o) * sqrtf(y0 * y0 + y1 * y1 + y2 * y2);
        if (lane < 16) {
          sYv[w][e0][0] = y0; sYv[w][e0][1] = y1; sYv[w][e0][2] = y2;
          sVny[w][e0] = vny;
        }
      }
      // ---- GEMM3: msg.out lin_scalar (16x129)@(129x128) + cutoff/elu + scatter
      float vnyR[8], cutR[8];
      int rowR[8], okR[8];
#pragma unroll
      for (int r8 = 0; r8 < 8; ++r8) {
        int m = rowhi + r8;
        vnyR[r8] = sVny[w][m]; cutR[r8] = sCut[w][m];
        rowR[r8] = sRow[w][m]; okR[r8] = sOk[w][m];
      }
      float gm[8] = {0.f, 0.f, 0.f, 0.f, 0.f, 0.f, 0.f, 0.f};
      for (int nt = 0; nt < 8; ++nt) {
        v8f acc = {0.f, 0.f, 0.f, 0.f, 0.f, 0.f, 0.f, 0.f};
#pragma unroll
        for (int kb = 0; kb < 4; ++kb)
          acc = wmma16(fragAs(A, kb, lane), fragBs(sWo, 8, kb, nt, lane), acc);
        int ncol = nt * 16 + ncb;
        float wv = (float)sWo[4 * 8 * 512 + ncol];
        float ob = cOB[ncol], gw = cGM[ncol];
        if (full) {  // fast path: no per-row EXEC masking
#pragma unroll
          for (int r8 = 0; r8 < 8; ++r8) {
            float val = acc[r8] + ob + vnyR[r8] * wv;
            gm[r8] += val * gw;
            atomicAdd(&a_sca[(size_t)rowR[r8] * 128 + ncol],
                      eluf(val * cutR[r8]));
          }
        } else {
#pragma unroll
          for (int r8 = 0; r8 < 8; ++r8) {
            float val = acc[r8] + ob + vnyR[r8] * wv;
            gm[r8] += val * gw;
            float ms = eluf(val * cutR[r8]);
            if (okR[r8]) atomicAdd(&a_sca[(size_t)rowR[r8] * 128 + ncol], ms);
          }
        }
      }
#pragma unroll
      for (int r8 = 0; r8 < 8; ++r8) {
        float v = gm[r8];
        v += __shfl_xor(v, 1); v += __shfl_xor(v, 2);
        v += __shfl_xor(v, 4); v += __shfl_xor(v, 8);
        if (ncb == 0) sGate[w][rowhi + r8] = sigm(v + gmb);
      }
      if (lane < 16 && (full || sOk[w][e0])) {
        float g = sGate[w][e0], cf = sCut[w][e0];
        int rr = sRow[w][e0];
#pragma unroll
        for (int d = 0; d < 3; ++d) {
          float mv = eluf(g * w2o * w1o * sYv[w][e0][d] * cf);
          atomicAdd(&a_vec[(size_t)rr * 3 + d], mv);
        }
      }
    }
    __syncthreads();
  }
}

// ---------------------------------------------------------------------------
// param flattening indices (insertion-order recursive flatten of setup_inputs)
enum {
  IN_XSCA = 0, IN_POS = 1, IN_EIDX = 2, IN_EATTR = 3,
  P1_LV = 4, P1_LV2 = 5, P1_GW = 6, P1_GB = 7, P1_LSW = 8, P1_LSB = 9, P1_VN = 10,
  MN_LV = 11, MN_LV2 = 12, MN_GW = 13, MN_GB = 14, MN_LSW = 15, MN_LSB = 16,
  ME_LV = 17, ME_LV2 = 18, ME_GW = 19, ME_GB = 20, ME_LSW = 21, ME_LSB = 22,
  M_VNRELU = 23, M_SCAW = 24, M_SCAB = 25, M_E2NW = 26, M_E2NB = 27,
  M_N2EW = 28, M_N2EB = 29, M_EVNW = 30,
  MO_LV = 31, MO_LV2 = 32, MO_GW = 33, MO_GB = 34, MO_LSW = 35, MO_LSB = 36,
  C_LV = 37, C_LV2 = 38, C_GW = 39, C_GB = 40, C_LSW = 41, C_LSB = 42,
  OT_LV = 43, OT_LV2 = 44, OT_GW = 45, OT_GB = 46, OT_LSW = 47, OT_LSB = 48,
  P_VEXP = 49, P_LNSW = 50, P_LNSB = 51, P_LNVW = 52, P_LNVB = 53, P_ACTVW = 54
};

extern "C" void kernel_launch(void* const* d_in, const int* in_sizes, int n_in,
                              void* d_out, int out_size, void* d_ws,
                              size_t ws_size, hipStream_t stream) {
  (void)n_in; (void)out_size; (void)ws_size;
  const float* x_sca = (const float*)d_in[IN_XSCA];
  const float* pos = (const float*)d_in[IN_POS];
  const int* eidx = (const int*)d_in[IN_EIDX];
  const float* eattr = (const float*)d_in[IN_EATTR];
  float* out = (float*)d_out;

  int nn = in_sizes[IN_XSCA] / 128;
  int ne = in_sizes[IN_EIDX] / 2;

  // ---- workspace carve-out (~131 MB) -------------------------------------
  float* f = (float*)d_ws;
  float* pre = f;    f += (size_t)nn * 128;
  float* s1 = f;     f += (size_t)nn * 128;  // reused as lns after norm
  float* ns = f;     f += (size_t)nn * 128;
  float* cs = f;     f += (size_t)nn * 128;
  float* a_sca = f;  f += (size_t)nn * 128;
  float* v1 = f;     f += (size_t)nn * 3;    // reused as lnv after norm
  float* nv = f;     f += (size_t)nn * 3;
  float* cv = f;     f += (size_t)nn * 3;
  float* a_vec = f;  f += (size_t)nn * 3;
  _Float16* h = (_Float16*)f;
  _Float16* wtPer1 = h; h += 4 * 8 * 512 + 128;
  _Float16* wtNode = h; h += 4 * 8 * 512 + 128;
  _Float16* wtCent = h; h += 4 * 8 * 512 + 128;
  _Float16* wtOT = h;   h += 4 * 8 * 512 + 128;
  _Float16* wtOut = h;  h += 4 * 8 * 512 + 128;
  _Float16* wtE1 = h;   h += 2 * 4 * 512 + 64;
  _Float16* wtSca = h;  h += 2 * 8 * 512;
  float* lns = s1;
  float* lnv = v1;

  // ---- pack weights to fragment-native f16 --------------------------------
  auto pack = [&](int idx, _Float16* dst, int N, int Kd, int vn) {
    int total = (Kd >> 5) * (N >> 4) * 512 + (vn ? N : 0);
    pack_wt_kernel<<<(total + 255) / 256, 256, 0, stream>>>(
        (const float*)d_in[idx], dst, N, Kd, vn);
  };
  pack(P1_LSW, wtPer1, 128, 128, 1);
  pack(MN_LSW, wtNode, 128, 128, 1);
  pack(C_LSW, wtCent, 128, 128, 1);
  pack(OT_LSW, wtOT, 128, 128, 1);
  pack(MO_LSW, wtOut, 128, 128, 1);
  pack(ME_LSW, wtE1, 64, 64, 1);
  pack(M_SCAW, wtSca, 128, 64, 0);

  zero_kernel<<<1024, 256, 0, stream>>>(a_sca, (long long)nn * 128);
  zero_kernel<<<64, 256, 0, stream>>>(a_vec, (long long)nn * 3);

  int nTilesN = (nn + 15) / 16;
  int gridN = (nTilesN + NW - 1) / NW;
  if (gridN > 2048) gridN = 2048;

  auto gemm = [&](const float* is, const float* iv, int w1, _Float16* wt,
                  int bias) {
    gvlin_gemm_kernel<<<gridN, 128, 0, stream>>>(
        is, iv, (const float*)d_in[w1], wt, (const float*)d_in[bias], pre, nn,
        nTilesN);
  };
  auto epi = [&](const float* iv, int w1, int w2, int gw, int gb, int vnw,
                 int mode, float* os, float* ov) {
    gvlin_epi_kernel<<<nn, 128, 0, stream>>>(
        pre, iv, (const float*)d_in[w1], (const float*)d_in[w2],
        (const float*)d_in[gw], (const float*)d_in[gb],
        (const float*)d_in[vnw], mode, os, ov, nn);
  };

  // per1: GVPerceptronVN on (x_sca, pos)
  gemm(x_sca, pos, P1_LV, wtPer1, P1_LSB);
  epi(pos, P1_LV, P1_LV2, P1_GW, P1_GB, P1_VN, 1, s1, v1);
  // msg.node gv_lin on (s1, v1) -> (ns, nv)
  gemm(s1, v1, MN_LV, wtNode, MN_LSB);
  epi(v1, MN_LV, MN_LV2, MN_GW, MN_GB, P1_VN, 0, ns, nv);
  // centroid gv_lin on (s1, v1) -> (cs, cv)
  gemm(s1, v1, C_LV, wtCent, C_LSB);
  epi(v1, C_LV, C_LV2, C_GW, C_GB, P1_VN, 0, cs, cv);

  // fused edge pipeline -> atomic accumulate (a_sca, a_vec)
  int nTilesE = (ne + 15) / 16;
  int gridE = (nTilesE + EW - 1) / EW;
  if (gridE > 4096) gridE = 4096;
  edge_fused_kernel<<<gridE, 64, 0, stream>>>(
      pos, eattr, eidx, ns, nv, wtE1, wtSca, wtOut,
      (const float*)d_in[ME_GW], (const float*)d_in[ME_GB],
      (const float*)d_in[ME_LSB], (const float*)d_in[M_SCAB],
      (const float*)d_in[M_E2NW], (const float*)d_in[M_E2NB],
      (const float*)d_in[M_N2EW], (const float*)d_in[M_N2EB],
      (const float*)d_in[MO_GW], (const float*)d_in[MO_GB],
      (const float*)d_in[MO_LSB], (const float*)d_in[ME_LV],
      (const float*)d_in[ME_LV2], (const float*)d_in[M_VNRELU],
      (const float*)d_in[M_EVNW], (const float*)d_in[MO_LV],
      (const float*)d_in[MO_LV2], (const float*)d_in[P_VEXP], a_sca, a_vec,
      ne, nTilesE);

  // layer norm + activations -> (lns, lnv)
  norm_act_kernel<<<nn, 128, 0, stream>>>(
      cs, cv, a_sca, a_vec, (const float*)d_in[P_LNSW],
      (const float*)d_in[P_LNSB], (const float*)d_in[P_LNVW],
      (const float*)d_in[P_LNVB], (const float*)d_in[P_ACTVW], lns, lnv, nn);

  // out_transform gv_lin -> d_out ([nn*128 scalars | nn*3 vectors])
  gemm(lns, lnv, OT_LV, wtOT, OT_LSB);
  epi(lnv, OT_LV, OT_LV2, OT_GW, OT_GB, P1_VN, 0, out, out + (size_t)nn * 128);
}
